// LSTMCRF_41248865911415
// MI455X (gfx1250) — matrix operand order repaired and checked by reading, not compiled, not run on hardware
//
#include <hip/hip_runtime.h>
#include <hip/hip_bf16.h>

typedef __attribute__((ext_vector_type(16))) _Float16 v16h;
typedef __attribute__((ext_vector_type(8)))  _Float16 v8h;
typedef __attribute__((ext_vector_type(8)))  float    v8f;

#define BN   64
#define SN   512
#define EN   256
#define HN   256
#define G4   1024     // 4*H
#define NTOK 32768    // S*B
#define NTAG 24

// ---------- helpers ----------

// Build the 16-bit A-matrix lane payload (v16h) for WMMA 16x16x32:
// lane holds two contiguous 8-halves chunks at +0 and +8 halves.
__device__ inline v16h load_a_tile(const _Float16* p) {
  v8h lo = *(const v8h*)(p);
  v8h hi = *(const v8h*)(p + 8);
  v16h r;
#pragma unroll
  for (int i = 0; i < 8; ++i) { r[i] = lo[i]; r[i + 8] = hi[i]; }
  return r;
}

__device__ inline v8f wmma_f16(v16h a, v16h b, v8f c) {
  return __builtin_amdgcn_wmma_f32_16x16x32_f16(false, a, false, b, (short)0, c,
                                                false, false);
}

__device__ inline float sigf(float x) { return 1.0f / (1.0f + __expf(-x)); }

// ---------- kernel 0: zero h-exchange buffers + barrier counters ----------
__global__ void init_state_kernel(_Float16* __restrict__ hbuf_f,
                                  _Float16* __restrict__ hbuf_b,
                                  unsigned* __restrict__ ctr) {
  size_t idx = (size_t)blockIdx.x * blockDim.x + threadIdx.x;  // < 2*16384
  hbuf_f[idx] = (_Float16)0.0f;
  hbuf_b[idx] = (_Float16)0.0f;
  if (idx < 2) ctr[idx] = 0u;
}

// ---------- kernel 1: cast weights to f16, pad Wout 24->32 rows ----------
__global__ void prep_weights_kernel(const float* __restrict__ Wih_f,
                                    const float* __restrict__ Whh_f,
                                    const float* __restrict__ Wih_b,
                                    const float* __restrict__ Whh_b,
                                    const float* __restrict__ Wout,
                                    _Float16* __restrict__ wih16f,
                                    _Float16* __restrict__ whh16f,
                                    _Float16* __restrict__ wih16b,
                                    _Float16* __restrict__ whh16b,
                                    _Float16* __restrict__ wout16) {
  const size_t WN = (size_t)G4 * EN;               // 262144
  size_t idx = (size_t)blockIdx.x * blockDim.x + threadIdx.x;
  if (idx < WN) {
    wih16f[idx] = (_Float16)Wih_f[idx];
  } else if (idx < 2 * WN) {
    size_t j = idx - WN;  whh16f[j] = (_Float16)Whh_f[j];
  } else if (idx < 3 * WN) {
    size_t j = idx - 2 * WN;  wih16b[j] = (_Float16)Wih_b[j];
  } else if (idx < 4 * WN) {
    size_t j = idx - 3 * WN;  whh16b[j] = (_Float16)Whh_b[j];
  } else if (idx < 4 * WN + 32 * 512) {
    size_t j = idx - 4 * WN;
    int n = (int)(j >> 9);
    int k = (int)(j & 511);
    wout16[j] = (n < NTAG) ? (_Float16)Wout[(size_t)n * 512 + k] : (_Float16)0.0f;
  }
}

// ---------- kernel 2: embedding gather, time-major, cast f16 ----------
__global__ void gather_embed_kernel(const int* __restrict__ x,
                                    const float* __restrict__ emb,
                                    _Float16* __restrict__ e16) {
  size_t idx = (size_t)blockIdx.x * blockDim.x + threadIdx.x;  // < NTOK*EN
  int k    = (int)(idx & (EN - 1));
  int ntok = (int)(idx >> 8);
  int s = ntok >> 6;          // / B
  int b = ntok & (BN - 1);
  int xv = x[(size_t)b * SN + s];
  e16[idx] = (_Float16)emb[(size_t)xv * EN + k];
}

// ---------- kernel 3: input projection GEMM  G = X @ W^T + (bih + bhh) ----------
__global__ void gemm_in_kernel(const _Float16* __restrict__ X,
                               const _Float16* __restrict__ W,
                               const float* __restrict__ bih,
                               const float* __restrict__ bhh,
                               float* __restrict__ G) {
  int wave  = (int)((blockIdx.x * blockDim.x + threadIdx.x) >> 5);
  int lane  = threadIdx.x & 31;
  int mTile = wave >> 4;        // 0..2047
  int nGrp  = wave & 15;        // 0..15 (64 cols each)
  int l16   = lane & 15;
  int half  = lane >> 4;
  int row   = mTile * 16 + l16;

  v8f acc[4];
#pragma unroll
  for (int g = 0; g < 4; ++g)
#pragma unroll
    for (int r = 0; r < 8; ++r) acc[g][r] = 0.0f;

  for (int kt = 0; kt < 8; ++kt) {
    int ka = kt * 32 + (half ? 16 : 0);
    v16h A = load_a_tile(X + (size_t)row * EN + ka);
#pragma unroll
    for (int g = 0; g < 4; ++g) {
      int n = nGrp * 64 + g * 16 + l16;
      const _Float16* bp = W + (size_t)n * EN + kt * 32 + (half ? 16 : 0);
      v16h Bv = *(const v16h*)bp;
      acc[g] = wmma_f16(A, Bv, acc[g]);
    }
  }
#pragma unroll
  for (int g = 0; g < 4; ++g) {
    int n = nGrp * 64 + g * 16 + l16;
    float bsum = bih[n] + bhh[n];
#pragma unroll
    for (int r = 0; r < 8; ++r) {
      int m = mTile * 16 + r + (half ? 8 : 0);
      G[(size_t)m * G4 + n] = acc[g][r] + bsum;
    }
  }
}

// ---------- kernel 4: LSTM recurrence, 4 workgroups per direction ----------
// Block `part` owns hidden units [part*64, part*64+64) => gate columns
// {g*256 + part*64 + u : g in 0..3, u in 0..63}.  Whh slice (128KB) lives in
// LDS; full h is exchanged through a double-buffered global buffer with a
// per-direction release/acquire spin barrier each step.
// LDS: wsl[256][256] f16 (128KB) + gl[64][256] f32 (64KB) + hl[64][256] f16 (32KB)
__global__ void __launch_bounds__(512)
lstm_rec_kernel(const float* __restrict__ Gf, const float* __restrict__ Gb,
                const _Float16* __restrict__ Wf, const _Float16* __restrict__ Wb,
                _Float16* __restrict__ hsf, _Float16* __restrict__ hsb,
                _Float16* __restrict__ hbuf_f, _Float16* __restrict__ hbuf_b,
                unsigned* __restrict__ ctr) {
  extern __shared__ char smem[];
  _Float16* wsl = (_Float16*)smem;                              // [256][256]
  float*    gl  = (float*)(smem + 131072);                      // [64][256]
  _Float16* hl  = (_Float16*)(smem + 131072 + 65536);           // [64][256]

  const int dir  = blockIdx.x >> 2;     // 0 fwd, 1 bwd
  const int part = blockIdx.x & 3;
  const float*    G    = dir ? Gb : Gf;
  const _Float16* W    = dir ? Wb : Wf;
  _Float16*       hs   = dir ? hsb : hsf;
  _Float16*       hbuf = dir ? hbuf_b : hbuf_f;   // 2 buffers of 16384 halves
  volatile unsigned* vc = ctr + dir;

  const int tid  = threadIdx.x;
  const int lane = tid & 31;
  const int wave = tid >> 5;    // 0..15
  const int l16  = lane & 15;
  const int half = lane >> 4;

  // Preload Whh slice: local row lr -> global row n = (lr>>6)*256 + part*64 + (lr&63)
  for (int i = tid; i < 8192; i += 512) {   // 8192 * 8 halves = 65536
    int idx8 = i * 8;
    int lr = idx8 >> 8;
    int k  = idx8 & 255;
    int n  = ((lr >> 6) << 8) + part * 64 + (lr & 63);
    *(v8h*)(wsl + idx8) = *(const v8h*)(W + (size_t)n * HN + k);
  }
  float c[8];
#pragma unroll
  for (int k = 0; k < 8; ++k) c[k] = 0.0f;
  __syncthreads();

  for (int step = 0; step < SN; ++step) {
    const int t  = dir ? (SN - 1 - step) : step;
    const int rd = step & 1;
    const int wr = (step + 1) & 1;

    // ---- fetch full h_prev (device-scope reads: bypass stale WGP$)
    {
      const volatile unsigned* src = (const volatile unsigned*)(hbuf + rd * 16384);
      unsigned* dst = (unsigned*)hl;
      for (int i = tid; i < 8192; i += 512) dst[i] = src[i];
      // prefetch this step's G rows while we wait
      __builtin_prefetch(G + (size_t)t * BN * G4 + tid * 32, 0, 0);
    }
    __syncthreads();

    // ---- phase 1: gl = h_prev @ Whh_slice^T   (64 tile jobs over 16 waves)
    for (int q = 0; q < 4; ++q) {
      int job = wave * 4 + q;
      int mt = job >> 4;        // 0..3 batch tile
      int qq = job & 15;        // g*4 + u
      int row = mt * 16 + l16;
      int lr  = qq * 16 + l16;  // local gate-column
      v8f acc;
#pragma unroll
      for (int r = 0; r < 8; ++r) acc[r] = 0.0f;
#pragma unroll
      for (int kt = 0; kt < 8; ++kt) {
        v16h A  = load_a_tile(hl + (size_t)row * HN + kt * 32 + (half ? 16 : 0));
        v16h Bv = *(const v16h*)(wsl + (size_t)lr * HN + kt * 32 + (half ? 16 : 0));
        acc = wmma_f16(A, Bv, acc);
      }
#pragma unroll
      for (int r = 0; r < 8; ++r) {
        int m = mt * 16 + r + (half ? 8 : 0);
        gl[(size_t)m * 256 + lr] = acc[r];
      }
    }
    __syncthreads();

    // ---- phase 2: gate math for our 64 units x 64 batches (8 per thread)
    const size_t base = (size_t)t * BN * G4;
#pragma unroll
    for (int k = 0; k < 8; ++k) {
      int e  = k * 512 + tid;
      int b  = e >> 6;            // batch
      int ju = e & 63;            // local unit
      int j  = part * 64 + ju;    // global unit
      size_t gb = base + (size_t)b * G4;
      size_t lb = (size_t)b * 256;
      float ig = gl[lb + ju]        + G[gb + j];
      float fg = gl[lb + 64 + ju]   + G[gb + 256 + j];
      float gg = gl[lb + 128 + ju]  + G[gb + 512 + j];
      float og = gl[lb + 192 + ju]  + G[gb + 768 + j];
      float ci = sigf(fg) * c[k] + sigf(ig) * tanhf(gg);
      c[k] = ci;
      float h = sigf(og) * tanhf(ci);
      _Float16 h16 = (_Float16)h;
      hs[(size_t)(t * BN + b) * HN + j] = h16;
      hbuf[wr * 16384 + b * 256 + j] = h16;
    }

    // ---- inter-workgroup barrier (release/acquire, per direction)
    __threadfence();          // push our h stores to device scope
    __syncthreads();          // all threads of block done + fenced
    if (tid == 0) {
      atomicAdd(ctr + dir, 1u);
      const unsigned target = 4u * (unsigned)(step + 1);
      while (*vc < target) { __builtin_amdgcn_s_sleep(1); }
      __threadfence();        // acquire
    }
    __syncthreads();
  }
}

// ---------- kernel 5: output projection  emis = [hf|hb] @ Wout^T + bout ----------
__global__ void gemm_out_kernel(const _Float16* __restrict__ hf,
                                const _Float16* __restrict__ hb,
                                const _Float16* __restrict__ Wout16,  // [32][512]
                                const float* __restrict__ bout,
                                float* __restrict__ emis) {           // [NTOK][24]
  int wave  = (int)((blockIdx.x * blockDim.x + threadIdx.x) >> 5);
  int lane  = threadIdx.x & 31;
  int mTile = wave >> 1;
  int nt    = wave & 1;
  int l16   = lane & 15;
  int half  = lane >> 4;
  int row   = mTile * 16 + l16;
  int n     = nt * 16 + l16;

  v8f acc;
#pragma unroll
  for (int r = 0; r < 8; ++r) acc[r] = 0.0f;

  for (int kt = 0; kt < 16; ++kt) {
    const _Float16* Xp = (kt < 8)
        ? hf + (size_t)row * HN + kt * 32
        : hb + (size_t)row * HN + (kt - 8) * 32;
    v16h A = load_a_tile(Xp + (half ? 16 : 0));
    const _Float16* bp = Wout16 + (size_t)n * 512 + kt * 32 + (half ? 16 : 0);
    v16h Bv = *(const v16h*)bp;
    acc = wmma_f16(A, Bv, acc);
  }
  if (n < NTAG) {
    float bo = bout[n];
#pragma unroll
    for (int r = 0; r < 8; ++r) {
      int m = mTile * 16 + r + (half ? 8 : 0);
      emis[(size_t)m * NTAG + n] = acc[r] + bo;
    }
  }
}

// ---------- kernel 6: Viterbi decode, one wave per batch element ----------
__global__ void viterbi_kernel(const float* __restrict__ emis,
                               const unsigned char* __restrict__ mask,
                               const float* __restrict__ trans,
                               const float* __restrict__ start,
                               const float* __restrict__ endv,
                               unsigned char* __restrict__ hist,  // [S][B][24]
                               float* __restrict__ out) {         // [B][S]
  int wave = (int)((blockIdx.x * blockDim.x + threadIdx.x) >> 5);
  int lane = threadIdx.x & 31;
  int b = wave;                       // 0..63
  int tag = lane;
  bool act = (lane < NTAG);
  int tg0 = act ? tag : 0;

  float tcol[NTAG];
#pragma unroll
  for (int p = 0; p < NTAG; ++p) tcol[p] = trans[(size_t)p * NTAG + tg0];

  float score = act ? (start[tag] + emis[(size_t)b * NTAG + tag]) : -1.0e30f;

  for (int s = 1; s < SN; ++s) {
    float et = emis[(size_t)(s * BN + b) * NTAG + tg0];
    float best = -1.0e30f;
    int bi = 0;
#pragma unroll
    for (int p = 0; p < NTAG; ++p) {
      float sp = __shfl(score, p, 32);
      float cand = sp + tcol[p];
      if (cand > best) { best = cand; bi = p; }
    }
    bool m = (mask[(size_t)b * SN + s] != 0);
    float ns = best + et;
    int idx = m ? bi : tag;
    if (m && act) score = ns;
    if (act) hist[(size_t)(s * BN + b) * NTAG + tag] = (unsigned char)idx;
  }

  float fin = act ? (score + endv[tag]) : -1.0e30f;
  float bv = -1.0e30f;
  int bt = 0;
#pragma unroll
  for (int p = 0; p < NTAG; ++p) {
    float v = __shfl(fin, p, 32);
    if (v > bv) { bv = v; bt = p; }
  }
  __syncthreads();   // wave's hist stores complete before lane-0 backtracks
  if (lane == 0) {
    int tg = bt;
    out[(size_t)b * SN + (SN - 1)] = (float)tg;
    for (int s = SN - 1; s >= 1; --s) {
      tg = hist[(size_t)(s * BN + b) * NTAG + tg];
      out[(size_t)b * SN + (s - 1)] = (float)tg;
    }
  }
}

// ---------- host launch ----------
extern "C" void kernel_launch(void* const* d_in, const int* in_sizes, int n_in,
                              void* d_out, int out_size, void* d_ws, size_t ws_size,
                              hipStream_t stream) {
  const int*   x      = (const int*)d_in[0];
  const unsigned char* mask = (const unsigned char*)d_in[1];  // bool -> u8
  const float* emb    = (const float*)d_in[2];
  const float* Wih_f  = (const float*)d_in[3];
  const float* Whh_f  = (const float*)d_in[4];
  const float* bih_f  = (const float*)d_in[5];
  const float* bhh_f  = (const float*)d_in[6];
  const float* Wih_b  = (const float*)d_in[7];
  const float* Whh_b  = (const float*)d_in[8];
  const float* bih_b  = (const float*)d_in[9];
  const float* bhh_b  = (const float*)d_in[10];
  const float* Wout   = (const float*)d_in[11];
  const float* bout   = (const float*)d_in[12];
  const float* trans  = (const float*)d_in[13];
  const float* start  = (const float*)d_in[14];
  const float* endv   = (const float*)d_in[15];
  float* out = (float*)d_out;

  char* ws = (char*)d_ws;
  // workspace layout (bytes)
  const size_t off_e16    = 0;                                   // 16 MB
  const size_t off_wih16f = off_e16    + (size_t)NTOK * EN * 2;
  const size_t off_wih16b = off_wih16f + (size_t)G4 * EN * 2;
  const size_t off_whh16f = off_wih16b + (size_t)G4 * EN * 2;
  const size_t off_whh16b = off_whh16f + (size_t)G4 * HN * 2;
  const size_t off_wout16 = off_whh16b + (size_t)G4 * HN * 2;
  const size_t off_Gf     = off_wout16 + (size_t)32 * 512 * 2;   // 134 MB
  const size_t off_Gb     = off_Gf     + (size_t)NTOK * G4 * 4;  // 134 MB
  const size_t off_hf16   = off_Gb     + (size_t)NTOK * G4 * 4;  // 16 MB
  const size_t off_hb16   = off_hf16   + (size_t)NTOK * HN * 2;  // 16 MB
  const size_t off_emis   = off_hb16   + (size_t)NTOK * HN * 2;  // 3 MB
  const size_t off_hist   = off_emis   + (size_t)NTOK * NTAG * 4;
  const size_t off_hbf    = off_hist   + (size_t)SN * BN * NTAG; // 64 KB
  const size_t off_hbb    = off_hbf    + (size_t)2 * BN * HN * 2;
  const size_t off_ctr    = off_hbb    + (size_t)2 * BN * HN * 2;

  _Float16* e16    = (_Float16*)(ws + off_e16);
  _Float16* wih16f = (_Float16*)(ws + off_wih16f);
  _Float16* wih16b = (_Float16*)(ws + off_wih16b);
  _Float16* whh16f = (_Float16*)(ws + off_whh16f);
  _Float16* whh16b = (_Float16*)(ws + off_whh16b);
  _Float16* wout16 = (_Float16*)(ws + off_wout16);
  float*    Gf     = (float*)(ws + off_Gf);
  float*    Gb     = (float*)(ws + off_Gb);
  _Float16* hf16   = (_Float16*)(ws + off_hf16);
  _Float16* hb16   = (_Float16*)(ws + off_hb16);
  float*    emis   = (float*)(ws + off_emis);
  unsigned char* hist = (unsigned char*)(ws + off_hist);
  _Float16* hbuf_f = (_Float16*)(ws + off_hbf);
  _Float16* hbuf_b = (_Float16*)(ws + off_hbb);
  unsigned* ctr    = (unsigned*)(ws + off_ctr);

  // 0. zero h-exchange buffers + barrier counters (2*16384 halves each)
  init_state_kernel<<<128, 256, 0, stream>>>(hbuf_f, hbuf_b, ctr);
  // 1. weight casts
  {
    size_t n = 4 * (size_t)G4 * EN + 32 * 512;
    prep_weights_kernel<<<(unsigned)((n + 255) / 256), 256, 0, stream>>>(
        Wih_f, Whh_f, Wih_b, Whh_b, Wout, wih16f, whh16f, wih16b, whh16b, wout16);
  }
  // 2. embedding gather + cast (time-major)
  gather_embed_kernel<<<(unsigned)((size_t)NTOK * EN / 256), 256, 0, stream>>>(
      x, emb, e16);
  // 3. input projections (both directions)
  gemm_in_kernel<<<4096, 256, 0, stream>>>(e16, wih16f, bih_f, bhh_f, Gf);
  gemm_in_kernel<<<4096, 256, 0, stream>>>(e16, wih16b, bih_b, bhh_b, Gb);
  // 4. recurrence: 8 blocks (2 dirs x 4 parts), 512 threads, 224KB dynamic LDS
  {
    size_t smem = 131072 + 65536 + 32768;  // wsl + gl + hl = 229376 B
    lstm_rec_kernel<<<8, 512, smem, stream>>>(Gf, Gb, whh16f, whh16b,
                                              hf16, hb16, hbuf_f, hbuf_b, ctr);
  }
  // 5. emission projection
  gemm_out_kernel<<<512, 256, 0, stream>>>(hf16, hb16, wout16, bout, emis);
  // 6. viterbi: 64 waves = 8 blocks x 256
  viterbi_kernel<<<8, 256, 0, stream>>>(emis, mask, trans, start, endv, hist, out);
}
